// STFT_10599979287127
// MI455X (gfx1250) — matrix operand order repaired and checked by reading, not compiled, not run on hardware
//
#include <hip/hip_runtime.h>

// Problem constants (fixed by the reference)
#define T_LEN   1048576
#define NBATCH  8
#define FILT    1024
#define HOP     256
#define NO      1026
#define NF      4097
#define NO_PAD  1152      // 9*128 zero-padded basis rows in workspace (no OOB reads)

typedef _Float16 v16h __attribute__((ext_vector_type(16)));
typedef _Float16 v8h  __attribute__((ext_vector_type(8)));
typedef float    v8f  __attribute__((ext_vector_type(8)));

__device__ __forceinline__ v16h frag16(const _Float16* p0, const _Float16* p1) {
  v8h a = *(const v8h*)p0;   // b128 load (ds or global depending on pointer)
  v8h b = *(const v8h*)p1;
  return __builtin_shufflevector(a,b,0,1,2,3,4,5,6,7,8,9,10,11,12,13,14,15);
}

#define WMMA(A,B,C) __builtin_amdgcn_wmma_f32_16x16x32_f16(false,(A),false,(B),(short)0,(C),false,false)

// ---------------------------------------------------------------------------
// Precompute: split basis f32 -> f16 hi + lo, row-major, zero-padded to 1152 rows
// ---------------------------------------------------------------------------
__global__ __launch_bounds__(256) void split_basis_kernel(
    const float* __restrict__ basis, _Float16* __restrict__ whi,
    _Float16* __restrict__ wlo)
{
  int idx = blockIdx.x * 256 + threadIdx.x;        // over NO_PAD*FILT
  if (idx >= NO_PAD * FILT) return;
  int o = idx >> 10;
  float v = (o < NO) ? basis[idx] : 0.0f;          // basis is [1026,1,1024] row-major
  _Float16 h = (_Float16)v;
  _Float16 l = (_Float16)(v - (float)h);
  whi[idx] = h;
  wlo[idx] = l;
}

// ---------------------------------------------------------------------------
// Main kernel: barrier-free k-loop, A from L2-resident split basis,
// B from LDS-staged split input span. 128 o x 48 f per block, 8 waves.
// Manually 2x-unrolled k-loop with ping-pong A fragments (no rotation copies).
// ---------------------------------------------------------------------------
__global__ __launch_bounds__(256) void stft_wmma_v3_kernel(
    const float* __restrict__ x, const _Float16* __restrict__ whi,
    const _Float16* __restrict__ wlo, float* __restrict__ out)
{
  constexpr int O_BLK  = 128;
  constexpr int F_BLK  = 48;
  constexpr int KC     = 32;
  constexpr int SPAN   = (F_BLK - 1) * HOP + FILT;   // 13056
  constexpr int SEGS   = SPAN / 256;                 // 51
  constexpr int XPITCH = 264;                        // 8-half pad -> bank spread
  constexpr int XHALFS = SEGS * XPITCH;              // 13464 halfs

  __shared__ __align__(16) _Float16 sXhi[XHALFS];    // 26928 B
  __shared__ __align__(16) _Float16 sXlo[XHALFS];    // 26928 B  (53856 B total)

  const int tid = threadIdx.x;
  const int b   = blockIdx.z;
  const int f0  = blockIdx.x * F_BLK;
  const int o0  = blockIdx.y * O_BLK;

  // ---- Stage input span (reflect-padded) once, split into f16 hi + lo ----
  const float* xb    = x + (size_t)b * T_LEN;
  const int    tbase = f0 * HOP - FILT / 2;
  if (tbase >= 0 && tbase + SPAN <= T_LEN) {
    // interior fast path: 8 samples per step, b128 in / b128 out
    for (int i = tid * 8; i < SPAN; i += 2048) {
      float4 v0 = *(const float4*)(xb + tbase + i);      // global_load_b128
      float4 v1 = *(const float4*)(xb + tbase + i + 4);
      _Float16 h0=(_Float16)v0.x, h1=(_Float16)v0.y, h2=(_Float16)v0.z, h3=(_Float16)v0.w;
      _Float16 h4=(_Float16)v1.x, h5=(_Float16)v1.y, h6=(_Float16)v1.z, h7=(_Float16)v1.w;
      v8h hh = { h0,h1,h2,h3,h4,h5,h6,h7 };
      v8h ll = { (_Float16)(v0.x-(float)h0), (_Float16)(v0.y-(float)h1),
                 (_Float16)(v0.z-(float)h2), (_Float16)(v0.w-(float)h3),
                 (_Float16)(v1.x-(float)h4), (_Float16)(v1.y-(float)h5),
                 (_Float16)(v1.z-(float)h6), (_Float16)(v1.w-(float)h7) };
      int a = (i >> 8) * XPITCH + (i & 255);
      *(v8h*)&sXhi[a] = hh;                               // ds_store_b128
      *(v8h*)&sXlo[a] = ll;
    }
  } else {
    // edge blocks (first / last f-block): per-element reflect
    for (int i = tid; i < SPAN; i += 256) {
      int g = tbase + i;
      if (g < 0)      g = -g;                             // np.pad reflect
      if (g >= T_LEN) g = 2 * T_LEN - 2 - g;
      float v = xb[g];
      _Float16 h = (_Float16)v;
      _Float16 l = (_Float16)(v - (float)h);
      int a = (i >> 8) * XPITCH + (i & 255);
      sXhi[a] = h;
      sXlo[a] = l;
    }
  }
  __syncthreads();    // the only barrier in the kernel

  const int lane = tid & 31;
  const int wv   = tid >> 5;      // wave id 0..7 -> o-tile
  const int mrow = lane & 15;
  const int hi16 = lane >> 4;     // half-wave selector

  // A fragment base: row-major split basis, ISA octet layout offsets
  const size_t arow = ((size_t)(o0 + wv * 16 + mrow) << 10) + hi16 * 8;
  const _Float16* aph = whi + arow;
  const _Float16* apl = wlo + arow;

  v8f acc[3];
  #pragma unroll
  for (int n = 0; n < 3; ++n) acc[n] = v8f{0.f,0.f,0.f,0.f,0.f,0.f,0.f,0.f};

  auto compute = [&](int k0, v16h Ah, v16h Al) {
    int t1 = k0 + hi16 * 8;                            // table-order K octet base
    int ka = (t1 >> 8) * XPITCH + (t1 & 255);          // k-dependent LDS offset
    #pragma unroll
    for (int n = 0; n < 3; ++n) {
      int ba = (n * 16 + mrow) * XPITCH + ka;
      v16h Bh = frag16(&sXhi[ba], &sXhi[ba + 16]);
      v16h Bl = frag16(&sXlo[ba], &sXlo[ba + 16]);
      // order chosen so Bl (then Al) die early -> shorter live ranges
      acc[n] = WMMA(Ah, Bl, acc[n]);
      acc[n] = WMMA(Al, Bh, acc[n]);
      acc[n] = WMMA(Ah, Bh, acc[n]);
    }
  };

  // ping-pong A fragments, manual 2x unroll: no register rotation copies
  v16h Ah0, Al0, Ah1, Al1;
  Ah0 = frag16(aph, aph + 16);
  Al0 = frag16(apl, apl + 16);
  for (int k0 = 0; k0 < FILT - 2 * KC; k0 += 2 * KC) {
    { const _Float16* p = aph + (k0 + KC); Ah1 = frag16(p, p + 16); }
    { const _Float16* p = apl + (k0 + KC); Al1 = frag16(p, p + 16); }
    compute(k0, Ah0, Al0);
    { const _Float16* p = aph + (k0 + 2 * KC); Ah0 = frag16(p, p + 16); }
    { const _Float16* p = apl + (k0 + 2 * KC); Al0 = frag16(p, p + 16); }
    compute(k0 + KC, Ah1, Al1);
  }
  { const _Float16* p = aph + (FILT - KC); Ah1 = frag16(p, p + 16); }
  { const _Float16* p = apl + (FILT - KC); Al1 = frag16(p, p + 16); }
  compute(FILT - 2 * KC, Ah0, Al0);
  compute(FILT - KC, Ah1, Al1);

  // ---- Store C tiles: VGPR r, lanes0-15 -> M=r, lanes16-31 -> M=r+8 ----
  const size_t obase = (size_t)b * NO * (size_t)NF;
  #pragma unroll
  for (int n = 0; n < 3; ++n) {
    int f = f0 + n * 16 + mrow;
    if (f < NF) {
      #pragma unroll
      for (int r = 0; r < 8; ++r) {
        int o = o0 + wv * 16 + hi16 * 8 + r;
        if (o < NO) out[obase + (size_t)o * NF + f] = acc[n][r];
      }
    }
  }
}

// ---------------------------------------------------------------------------
// Fallback (no workspace): self-contained version with per-chunk W staging
// ---------------------------------------------------------------------------
__global__ __launch_bounds__(256) void stft_wmma_fb_kernel(
    const float* __restrict__ x, const float* __restrict__ basis,
    float* __restrict__ out)
{
  constexpr int O_BLK  = 128;
  constexpr int F_BLK  = 32;
  constexpr int KC     = 32;
  constexpr int SPAN   = (F_BLK - 1) * HOP + FILT;   // 8960
  constexpr int SEGS   = SPAN / 256;                 // 35
  constexpr int XPITCH = 264;
  constexpr int XHALFS = SEGS * XPITCH;              // 9240
  constexpr int WPITCH = 40;
  constexpr int WHALFS = O_BLK * WPITCH;             // 5120

  __shared__ __align__(16) _Float16 sXhi[XHALFS];
  __shared__ __align__(16) _Float16 sXlo[XHALFS];
  __shared__ __align__(16) _Float16 sWhi[WHALFS];
  __shared__ __align__(16) _Float16 sWlo[WHALFS];

  const int tid = threadIdx.x;
  const int b   = blockIdx.z;
  const int f0  = blockIdx.x * F_BLK;
  const int o0  = blockIdx.y * O_BLK;

  const float* xb    = x + (size_t)b * T_LEN;
  const int    tbase = f0 * HOP - FILT / 2;
  for (int i = tid; i < SPAN; i += 256) {
    int g = tbase + i;
    if (g < 0)      g = -g;
    if (g >= T_LEN) g = 2 * T_LEN - 2 - g;
    float v = xb[g];
    _Float16 h = (_Float16)v;
    _Float16 l = (_Float16)(v - (float)h);
    int a = (i >> 8) * XPITCH + (i & 255);
    sXhi[a] = h; sXlo[a] = l;
  }

  const int lane = tid & 31;
  const int wv   = tid >> 5;
  const int mrow = lane & 15;
  const int hi16 = lane >> 4;

  v8f acc0 = {0.f,0.f,0.f,0.f,0.f,0.f,0.f,0.f};
  v8f acc1 = {0.f,0.f,0.f,0.f,0.f,0.f,0.f,0.f};

  const int wr  = tid >> 1;
  const int wjb = (tid & 1) * 16;
  const int wo  = o0 + wr;

  for (int k0 = 0; k0 < FILT; k0 += KC) {
    __syncthreads();
    {
      float vv[16];
      if (wo < NO) {
        const float* wp = basis + (size_t)wo * FILT + k0 + wjb;
        #pragma unroll
        for (int q = 0; q < 4; ++q) {
          float4 f4 = *(const float4*)(wp + 4 * q);
          vv[4*q+0]=f4.x; vv[4*q+1]=f4.y; vv[4*q+2]=f4.z; vv[4*q+3]=f4.w;
        }
      } else {
        #pragma unroll
        for (int q = 0; q < 16; ++q) vv[q] = 0.0f;
      }
      #pragma unroll
      for (int q = 0; q < 16; ++q) {
        _Float16 h = (_Float16)vv[q];
        _Float16 l = (_Float16)(vv[q] - (float)h);
        sWhi[wr * WPITCH + wjb + q] = h;
        sWlo[wr * WPITCH + wjb + q] = l;
      }
    }
    __syncthreads();

    const _Float16* ap  = &sWhi[(wv * 16 + mrow) * WPITCH + hi16 * 8];
    const _Float16* alp = &sWlo[(wv * 16 + mrow) * WPITCH + hi16 * 8];
    v16h Ah = frag16(ap,  ap  + 16);
    v16h Al = frag16(alp, alp + 16);

    #pragma unroll
    for (int n = 0; n < 2; ++n) {
      int nfl = n * 16 + mrow;
      int t1  = k0 + hi16 * 8;
      int ba  = (nfl + (t1 >> 8)) * XPITCH + (t1 & 255);
      v16h Bh = frag16(&sXhi[ba], &sXhi[ba + 16]);
      v16h Bl = frag16(&sXlo[ba], &sXlo[ba + 16]);
      v8f* acc = n ? &acc1 : &acc0;
      *acc = WMMA(Ah, Bl, *acc);
      *acc = WMMA(Al, Bh, *acc);
      *acc = WMMA(Ah, Bh, *acc);
    }
  }

  const size_t obase = (size_t)b * NO * (size_t)NF;
  #pragma unroll
  for (int n = 0; n < 2; ++n) {
    const v8f& acc = n ? acc1 : acc0;
    int f = f0 + n * 16 + mrow;
    if (f < NF) {
      #pragma unroll
      for (int r = 0; r < 8; ++r) {
        int o = o0 + wv * 16 + hi16 * 8 + r;
        if (o < NO) out[obase + (size_t)o * NF + f] = acc[r];
      }
    }
  }
}

extern "C" void kernel_launch(void* const* d_in, const int* in_sizes, int n_in,
                              void* d_out, int out_size, void* d_ws, size_t ws_size,
                              hipStream_t stream) {
  (void)in_sizes; (void)n_in; (void)out_size;
  const float* x     = (const float*)d_in[0];   // [8, 1048576] f32
  const float* basis = (const float*)d_in[1];   // [1026, 1, 1024] f32
  float*       out   = (float*)d_out;           // [8, 1026, 4097] f32

  const size_t ws_needed = (size_t)2 * NO_PAD * FILT * sizeof(_Float16); // 4.7 MB
  if (d_ws != nullptr && ws_size >= ws_needed) {
    _Float16* whi = (_Float16*)d_ws;
    _Float16* wlo = whi + (size_t)NO_PAD * FILT;
    split_basis_kernel<<<dim3((NO_PAD * FILT) / 256), dim3(256), 0, stream>>>(
        basis, whi, wlo);
    dim3 grid((NF + 47) / 48, (NO + 127) / 128, NBATCH);   // 86 x 9 x 8
    stft_wmma_v3_kernel<<<grid, dim3(256), 0, stream>>>(x, whi, wlo, out);
  } else {
    dim3 grid((NF + 31) / 32, (NO + 127) / 128, NBATCH);   // 129 x 9 x 8
    stft_wmma_fb_kernel<<<grid, dim3(256), 0, stream>>>(x, basis, out);
  }
}
